// Cluster_1932735283321
// MI455X (gfx1250) — compile-verified, matrix operand order checked by reading, not run on hardware
//
#include <hip/hip_runtime.h>

typedef __attribute__((ext_vector_type(16))) __bf16 v16bf;
typedef __attribute__((ext_vector_type(8)))  __bf16 v8bf;
typedef __attribute__((ext_vector_type(8)))  float  v8f;

#define B_   8
#define P_   784
#define D_   768
#define PD_  128
#define NC_  512
#define NB_  8192
#define BP_  (B_ * P_)
#define KT_D  (D_ / 32)    // 24 WMMA k-steps for K=768
#define KT_PD (PD_ / 32)   // 4 WMMA k-steps for K=128
#define INV_TEMP 14.285714285714286f   // 1/0.07
#define POS_TH 0.5f
#define NEG_TH 0.1f
#define SHIFT  15.0f       // |cs/T| <= ~14.6, so fixed-shift LSE is overflow-safe

// ---------------------------------------------------------------------------
// Fragment loader: two 16-byte chunks -> one 16xbf16 fragment.
// ISA 16-bit 16x32 layout: lane&15 = row(M)/col(N); lane>>4 picks K-halves
// {0..7,16..23} (kh=0) vs {8..15,24..31} (kh=1). p points at (kb + kh*8).
// ---------------------------------------------------------------------------
__device__ __forceinline__ v16bf load_frag16(const __bf16* p) {
    v8bf lo = *reinterpret_cast<const v8bf*>(p);
    v8bf hi = *reinterpret_cast<const v8bf*>(p + 16);
    return __builtin_shufflevector(lo, hi, 0,1,2,3,4,5,6,7,8,9,10,11,12,13,14,15);
}

__device__ __forceinline__ v8f wmma_bf16(v16bf a, v16bf b, v8f c) {
    return __builtin_amdgcn_wmma_f32_16x16x32_bf16(
        /*neg_a=*/false, a, /*neg_b=*/false, b,
        /*c_mod=*/(short)0, c, /*reuse_a=*/false, /*reuse_b=*/false);
}

// ---------------------------------------------------------------------------
// 1) Row L2-normalize f32 -> bf16.  One wave per row; grid = rows/8, block 256.
// ---------------------------------------------------------------------------
__global__ void k_row_l2norm(const float* __restrict__ in, __bf16* __restrict__ out,
                             int rows, int cols) {
    const int wave = threadIdx.x >> 5, lane = threadIdx.x & 31;
    const int row = blockIdx.x * 8 + wave;
    if (row >= rows) return;
    const float* src = in + (size_t)row * cols;
    float s = 0.f;
    for (int c = lane; c < cols; c += 32) { float v = src[c]; s += v * v; }
    for (int off = 16; off; off >>= 1) s += __shfl_xor(s, off);
    const float scale = 1.0f / fmaxf(sqrtf(s), 1e-12f);
    __bf16* dst = out + (size_t)row * cols;
    for (int c = lane; c < cols; c += 32) dst[c] = (__bf16)(src[c] * scale);
}

// ---------------------------------------------------------------------------
// 2) dist = norm_feat . norm_cbk^T (K=768) + per-row argmax over 512 codes.
// Register-resident A fragments; 1-deep pipelined B loads.
// ---------------------------------------------------------------------------
__global__ void __launch_bounds__(256, 1)
k_dist_argmax(const __bf16* __restrict__ nFeat,
              const __bf16* __restrict__ nCbk,
              int* __restrict__ ind) {
    __shared__ float rBV[8][16];
    __shared__ int   rBI[8][16];

    const int rowBase = blockIdx.x * 16;
    const int wave = threadIdx.x >> 5, lane = threadIdx.x & 31;
    const int m = lane & 15, kh = lane >> 4;

    // A fragments resident in VGPRs for the whole code loop
    const __bf16* aP = nFeat + (size_t)(rowBase + m) * D_ + kh * 8;
    v16bf aF[KT_D];
#pragma unroll
    for (int kb = 0; kb < KT_D; ++kb) aF[kb] = load_frag16(aP + kb * 32);

    float bestV[8]; int bestI[8];
#pragma unroll
    for (int j = 0; j < 8; ++j) { bestV[j] = -3.0e38f; bestI[j] = 0; }

    for (int qt = wave; qt < NC_ / 16; qt += 8) {
        const int qBase = qt * 16;
        const __bf16* bP = nCbk + (size_t)(qBase + m) * D_ + kh * 8;
        v8f acc = {};
        v16bf bcur = load_frag16(bP);
#pragma unroll
        for (int kb = 0; kb < KT_D - 1; ++kb) {
            v16bf bnext = load_frag16(bP + (kb + 1) * 32);
            acc = wmma_bf16(aF[kb], bcur, acc);
            bcur = bnext;
        }
        acc = wmma_bf16(aF[KT_D - 1], bcur, acc);

        const int col = qBase + m;
#pragma unroll
        for (int j = 0; j < 8; ++j) {
            float v = acc[j];
            if (v > bestV[j]) { bestV[j] = v; bestI[j] = col; }
        }
    }
    // reduce across the 16 lanes that share a row group
#pragma unroll
    for (int j = 0; j < 8; ++j) {
        float v = bestV[j]; int i = bestI[j];
        for (int off = 1; off < 16; off <<= 1) {
            float ov = __shfl_xor(v, off); int oi = __shfl_xor(i, off);
            if (ov > v || (ov == v && oi < i)) { v = ov; i = oi; }
        }
        bestV[j] = v; bestI[j] = i;
    }
    if ((lane & 15) == 0) {
#pragma unroll
        for (int j = 0; j < 8; ++j) { int r = kh * 8 + j; rBV[wave][r] = bestV[j]; rBI[wave][r] = bestI[j]; }
    }
    __syncthreads();
    if (threadIdx.x < 16) {
        const int r = threadIdx.x;
        float v = rBV[0][r]; int i = rBI[0][r];
        for (int w = 1; w < 8; ++w) {
            float ov = rBV[w][r]; int oi = rBI[w][r];
            if (ov > v || (ov == v && oi < i)) { v = ov; i = oi; }
        }
        ind[rowBase + r] = i;
    }
}

// ---------------------------------------------------------------------------
// 3) norm_vq[i] = norm_cbk[ind[i]]   (l2norm(codebook[ind]) == norm_cbk[ind])
// ---------------------------------------------------------------------------
__global__ void k_gather_vq(const int* __restrict__ ind,
                            const __bf16* __restrict__ nCbk,
                            __bf16* __restrict__ nVq) {
    const int row = blockIdx.x;
    const int c = ind[row];
    const unsigned* s = reinterpret_cast<const unsigned*>(nCbk + (size_t)c * D_);
    unsigned* d = reinterpret_cast<unsigned*>(nVq + (size_t)row * D_);
    for (int i = threadIdx.x; i < D_ / 2; i += blockDim.x) d[i] = s[i];
}

// ---------------------------------------------------------------------------
// 4) Fused dual-GEMM masked-NCE, fixed-shift log-sum-exp.
//    Block = 16 anchors; A fragments (24 vq + 4 proj) live in VGPRs for the
//    whole q loop; B fragment loads are 1-deep software pipelined; next
//    q-tile is prefetched with global_prefetch.
// ---------------------------------------------------------------------------
__global__ void __launch_bounds__(256, 1)
k_nce(const __bf16* __restrict__ aVq,   // [BP,768]
      const __bf16* __restrict__ aPr,   // [BP,128]
      const __bf16* __restrict__ qVq,   // [Q,768]
      const __bf16* __restrict__ qPr,   // [Q,128]
      int Q, int selfExclude,
      float* __restrict__ numAcc,       // [8]
      float* __restrict__ cntAcc) {     // [8]
    __shared__ float rS[8][16], rA[8][16], rC[8][16];

    const int rowBase = blockIdx.x * 16;
    const int wave = threadIdx.x >> 5, lane = threadIdx.x & 31;
    const int m = lane & 15, kh = lane >> 4;
    const int Qt = Q / 16;

    // --- register-resident A fragments (invariant over the q loop) ---
    const __bf16* aVqp = aVq + (size_t)(rowBase + m) * D_ + kh * 8;
    v16bf aVqF[KT_D];
#pragma unroll
    for (int kb = 0; kb < KT_D; ++kb) aVqF[kb] = load_frag16(aVqp + kb * 32);

    const __bf16* aPrp = aPr + (size_t)(rowBase + m) * PD_ + kh * 8;
    v16bf aPrF[KT_PD];
#pragma unroll
    for (int k4 = 0; k4 < KT_PD; ++k4) aPrF[k4] = load_frag16(aPrp + k4 * 32);

    float sRun[8], aRun[8], cRun[8];
#pragma unroll
    for (int j = 0; j < 8; ++j) { sRun[j] = 0.f; aRun[j] = 0.f; cRun[j] = 0.f; }

    for (int qt = wave; qt < Qt; qt += 8) {
        const int qBase = qt * 16;
        const __bf16* bVqp = qVq + (size_t)(qBase + m) * D_ + kh * 8;
        const __bf16* bPrp = qPr + (size_t)(qBase + m) * PD_ + kh * 8;

        // prefetch this wave's next q-tile (stride = 8 tiles = 128 rows)
        if (qt + 8 < Qt) {
            __builtin_prefetch(bVqp + (size_t)128 * D_, 0, 0);
            __builtin_prefetch(bPrp + (size_t)128 * PD_, 0, 0);
        }

        // mask GEMM: K=768, pipelined B
        v8f accM = {};
        v16bf bcur = load_frag16(bVqp);
#pragma unroll
        for (int kb = 0; kb < KT_D - 1; ++kb) {
            v16bf bnext = load_frag16(bVqp + (kb + 1) * 32);
            accM = wmma_bf16(aVqF[kb], bcur, accM);
            bcur = bnext;
        }
        accM = wmma_bf16(aVqF[KT_D - 1], bcur, accM);

        // similarity GEMM: K=128, all B frags up front
        v16bf bPrFr[KT_PD];
#pragma unroll
        for (int k4 = 0; k4 < KT_PD; ++k4) bPrFr[k4] = load_frag16(bPrp + k4 * 32);
        v8f accS = {};
#pragma unroll
        for (int k4 = 0; k4 < KT_PD; ++k4) accS = wmma_bf16(aPrF[k4], bPrFr[k4], accS);

        const int col = qBase + m;
#pragma unroll
        for (int j = 0; j < 8; ++j) {
            const float cbk = accM[j];
            const float cs  = accS[j] * INV_TEMP;
            const int row   = rowBase + kh * 8 + j;
            const bool pos  = (cbk > POS_TH) && !(selfExclude && (col == row));
            const bool neg  = (cbk < NEG_TH);
            const float e   = __expf(cs - SHIFT);        // bounded: no max needed
            if (pos || neg) sRun[j] += e;
            if (pos) { aRun[j] += cs; cRun[j] += 1.f; }
        }
    }

    // plain-sum reductions across the 16 lanes sharing each row group
#pragma unroll
    for (int j = 0; j < 8; ++j) {
        float ss = sRun[j], aa = aRun[j], cc = cRun[j];
        for (int off = 1; off < 16; off <<= 1) {
            ss += __shfl_xor(ss, off);
            aa += __shfl_xor(aa, off);
            cc += __shfl_xor(cc, off);
        }
        sRun[j] = ss; aRun[j] = aa; cRun[j] = cc;
    }
    if ((lane & 15) == 0) {
#pragma unroll
        for (int j = 0; j < 8; ++j) {
            int r = kh * 8 + j;
            rS[wave][r] = sRun[j]; rA[wave][r] = aRun[j]; rC[wave][r] = cRun[j];
        }
    }
    __syncthreads();
    if (threadIdx.x < 16) {
        const int r = threadIdx.x;
        float ss = rS[0][r], aa = rA[0][r], cc = rC[0][r];
        for (int w = 1; w < 8; ++w) { ss += rS[w][r]; aa += rA[w][r]; cc += rC[w][r]; }
        if (cc > 0.f) {
            // sum over positives of (SHIFT - cs) + c*log(sum_masked e^{cs-SHIFT})
            const float contrib = cc * SHIFT - aa + cc * __logf(ss);
            const int b = (rowBase + r) / P_;
            atomicAdd(&numAcc[b], contrib);
            atomicAdd(&cntAcc[b], cc);
        }
    }
}

// ---------------------------------------------------------------------------
// 5) init / finalize
// ---------------------------------------------------------------------------
__global__ void k_init(float* acc) { if (threadIdx.x < 32) acc[threadIdx.x] = 0.f; }

__global__ void k_finalize(const float* __restrict__ acc, float* __restrict__ out) {
    if (threadIdx.x == 0) {
        float t = 0.f;
        for (int b = 0; b < B_; ++b) {
            t += acc[b]      / acc[8 + b];    // in-batch
            t += acc[16 + b] / acc[24 + b];   // bank
        }
        out[0] = t / (2.0f * B_);
    }
}

// ---------------------------------------------------------------------------
// launch
// ---------------------------------------------------------------------------
extern "C" void kernel_launch(void* const* d_in, const int* in_sizes, int n_in,
                              void* d_out, int out_size, void* d_ws, size_t ws_size,
                              hipStream_t stream) {
    (void)in_sizes; (void)n_in; (void)out_size; (void)ws_size;
    const float* feat     = (const float*)d_in[0];   // [8,784,768]
    const float* proj     = (const float*)d_in[1];   // [8,784,128]
    const float* projEma  = (const float*)d_in[2];   // [8,784,128]
    const float* codebook = (const float*)d_in[3];   // [512,768]
    const float* bankVq   = (const float*)d_in[4];   // [8192,768]
    const float* bankEma  = (const float*)d_in[5];   // [8192,128]
    float* out = (float*)d_out;

    char* ws = (char*)d_ws;
    size_t off = 0;
    auto take = [&](size_t bytes) { char* p = ws + off; off = (off + bytes + 255) & ~(size_t)255; return p; };
    __bf16* nFeat    = (__bf16*)take((size_t)BP_ * D_  * 2);
    __bf16* nCbk     = (__bf16*)take((size_t)NC_ * D_  * 2);
    __bf16* nProj    = (__bf16*)take((size_t)BP_ * PD_ * 2);
    __bf16* nEma     = (__bf16*)take((size_t)BP_ * PD_ * 2);
    __bf16* nBankVq  = (__bf16*)take((size_t)NB_ * D_  * 2);
    __bf16* nBankEma = (__bf16*)take((size_t)NB_ * PD_ * 2);
    __bf16* nVq      = (__bf16*)take((size_t)BP_ * D_  * 2);
    int*    ind      = (int*)   take((size_t)BP_ * 4);
    float*  acc      = (float*) take(32 * 4);   // [num0|cnt0|num1|cnt1] x 8

    k_init<<<1, 32, 0, stream>>>(acc);

    k_row_l2norm<<<BP_ / 8, 256, 0, stream>>>(feat,     nFeat,    BP_, D_);
    k_row_l2norm<<<NC_ / 8, 256, 0, stream>>>(codebook, nCbk,     NC_, D_);
    k_row_l2norm<<<BP_ / 8, 256, 0, stream>>>(proj,     nProj,    BP_, PD_);
    k_row_l2norm<<<BP_ / 8, 256, 0, stream>>>(projEma,  nEma,     BP_, PD_);
    k_row_l2norm<<<NB_ / 8, 256, 0, stream>>>(bankVq,   nBankVq,  NB_, D_);
    k_row_l2norm<<<NB_ / 8, 256, 0, stream>>>(bankEma,  nBankEma, NB_, PD_);

    k_dist_argmax<<<BP_ / 16, 256, 0, stream>>>(nFeat, nCbk, ind);
    k_gather_vq<<<BP_, 256, 0, stream>>>(ind, nCbk, nVq);

    // in-batch pass: Q = BP, self-exclusion on
    k_nce<<<BP_ / 16, 256, 0, stream>>>(nVq, nProj, nVq, nEma, BP_, 1, acc + 0, acc + 8);
    // bank pass: Q = NB
    k_nce<<<BP_ / 16, 256, 0, stream>>>(nVq, nProj, nBankVq, nBankEma, NB_, 0, acc + 16, acc + 24);

    k_finalize<<<1, 64, 0, stream>>>(acc, out);
}
